// NeuralODE_87290915324511
// MI455X (gfx1250) — compile-verified, matrix operand order checked by reading, not run on hardware
//
#include <hip/hip_runtime.h>
#include <hip/hip_bf16.h>

typedef __bf16 bf16_t;
typedef __attribute__((ext_vector_type(16))) bf16_t v16bf;
typedef __attribute__((ext_vector_type(8)))  float  v8f;

#define NWG    64
#define BLOCK  256
#define NWAVE  8

#define DIM  1024
#define HID  4096
#define NSTAGE 6

// ---- Tsit5 tableau (f32) ----
__constant__ float c_A[6][5] = {
  {0.f, 0.f, 0.f, 0.f, 0.f},
  {0.161f, 0.f, 0.f, 0.f, 0.f},
  {-0.008480655492356989f, 0.335480655492357f, 0.f, 0.f, 0.f},
  {2.8971530571054935f, -6.359448489975075f, 4.3622954328695815f, 0.f, 0.f},
  {5.325864828439257f, -11.748883564062828f, 7.4955393428898365f, -0.09249506636175525f, 0.f},
  {5.86145544294642f, -12.92096931784711f, 8.159367898576159f, -0.071584973281401f, -0.028269050394068383f}
};
#define TB1 0.09646076681806523f
#define TB2 0.01f
#define TB3 0.4798896504144996f
#define TB4 1.379008574103742f
#define TB5 (-3.290069515436081f)
#define TB6 2.324710524099774f

// ---- workspace layout (bytes) ----
constexpr size_t SZ_W1  = (size_t)HID * DIM * 2;   // 8 MB bf16
constexpr size_t SZ_W2  = (size_t)HID * HID * 2;   // 32 MB bf16
constexpr size_t SZ_W3  = (size_t)DIM * HID * 2;   // 8 MB bf16
constexpr size_t OFF_W1 = 0;
constexpr size_t OFF_W2 = OFF_W1 + SZ_W1;
constexpr size_t OFF_W3 = OFF_W2 + SZ_W2;
constexpr size_t OFF_Y  = OFF_W3 + SZ_W3;
constexpr size_t OFF_H1 = OFF_Y  + 4096;           // DIM f32 (padded)
constexpr size_t OFF_H2 = OFF_H1 + (size_t)HID * 4;
constexpr size_t OFF_K  = OFF_H2 + (size_t)HID * 4;
constexpr size_t OFF_BAR = (OFF_K + (size_t)NSTAGE * DIM * 4 + 255) & ~(size_t)255;

// =====================================================================
// Pack f32 row-major weights into WMMA-native tiled bf16 layout.
// Tile = 16 rows x 32 K. Block b = tr*nkb + kb holds 32 lanes x 16 halves:
//   lane l (m = l&15, g = (l&16)?8:0) half j -> W[tr*16+m][kb*32 + g + j + (j>=8?8:0)]
// =====================================================================
__global__ __launch_bounds__(256) void packW(const float* __restrict__ src,
                                             bf16_t* __restrict__ dst,
                                             int R, int C) {
  int p = blockIdx.x * 256 + threadIdx.x;
  int n = R * C;
  if (p >= n) return;
  int j  = p & 15;
  int l  = (p >> 4) & 31;
  int b  = p >> 9;
  int nkb = C >> 5;
  int tr = b / nkb;
  int kb = b - tr * nkb;
  int row = tr * 16 + (l & 15);
  int g   = (l & 16) ? 8 : 0;
  int col = kb * 32 + g + j + ((j >= 8) ? 8 : 0);
  dst[p] = (bf16_t)src[(size_t)row * C + col];
}

// =====================================================================
// Device-wide generation barrier (single persistent kernel, NWG blocks).
// =====================================================================
__device__ __forceinline__ void grid_barrier(unsigned* cnt, unsigned* gen) {
  __syncthreads();
  if (threadIdx.x == 0) {
    __threadfence();
    unsigned g = __hip_atomic_load(gen, __ATOMIC_RELAXED, __HIP_MEMORY_SCOPE_AGENT);
    unsigned a = __hip_atomic_fetch_add(cnt, 1u, __ATOMIC_ACQ_REL, __HIP_MEMORY_SCOPE_AGENT);
    if (a == (unsigned)(NWG - 1)) {
      __hip_atomic_store(cnt, 0u, __ATOMIC_RELAXED, __HIP_MEMORY_SCOPE_AGENT);
      __hip_atomic_store(gen, g + 1u, __ATOMIC_RELEASE, __HIP_MEMORY_SCOPE_AGENT);
    } else {
      while (__hip_atomic_load(gen, __ATOMIC_ACQUIRE, __HIP_MEMORY_SCOPE_AGENT) == g) {
        __builtin_amdgcn_s_sleep(2);
      }
    }
  }
  __syncthreads();
}

// =====================================================================
// One matvec layer: out[Rout] = (tanh?) (Wpk[Rout x K] @ vec + bias)
// Tiles of 16 rows distributed over WGs; 8 waves split K; LDS reduction.
// B operand: all 16 columns replicated with the input vector -> column 0
// of the 16x16 f32 result (lanes 0 and 16) carries the matvec outputs.
// =====================================================================
__device__ __forceinline__ void matvec_layer(const bf16_t* __restrict__ Wpk,
                                             const float* __restrict__ bias,
                                             const bf16_t* vecLds,
                                             float* redLds,
                                             float* __restrict__ out,
                                             int Rout, int K, bool doTanh) {
  const int tid  = threadIdx.x;
  const int lane = tid & 31;
  const int wave = tid >> 5;
  const int g    = (lane & 16) ? 8 : 0;
  const int nkb  = K >> 5;            // 16x32 K-blocks per row tile
  const int kbW  = nkb >> 3;          // K-blocks per wave (8 waves)
  const int kb0  = wave * kbW;
  const int ntiles = Rout >> 4;

  for (int t = blockIdx.x; t < ntiles; t += NWG) {
    const bf16_t* base = Wpk + (((size_t)t * nkb + kb0) << 9) + (lane << 4);
    if (t + NWG < ntiles)  // warm next tile of this wave into cache
      __builtin_prefetch(Wpk + (((size_t)(t + NWG) * nkb + kb0) << 9) + (lane << 4), 0, 3);

    v8f acc0 = {};
    v8f acc1 = {};
    for (int i = 0; i < kbW; i += 2) {  // kbW is 4 or 16 -> always even
      const int kA = kb0 + i;
      union { v16bf v; uint4 q[2]; } b0, b1;
      v16bf a0 = *(const v16bf*)(base + ((size_t)i << 9));
      b0.q[0] = *(const uint4*)(vecLds + (kA << 5) + g);
      b0.q[1] = *(const uint4*)(vecLds + (kA << 5) + 16 + g);
      acc0 = __builtin_amdgcn_wmma_f32_16x16x32_bf16(false, a0, false, b0.v,
                                                     (short)0, acc0, false, false);
      v16bf a1 = *(const v16bf*)(base + ((size_t)(i + 1) << 9));
      b1.q[0] = *(const uint4*)(vecLds + ((kA + 1) << 5) + g);
      b1.q[1] = *(const uint4*)(vecLds + ((kA + 1) << 5) + 16 + g);
      acc1 = __builtin_amdgcn_wmma_f32_16x16x32_bf16(false, a1, false, b1.v,
                                                     (short)0, acc1, false, false);
    }
    acc0 += acc1;

    // column 0 lives in lane 0 (rows 0..7) and lane 16 (rows 8..15)
    if ((lane & 15) == 0) {
      float* dst = redLds + wave * 16 + ((lane >> 4) << 3);
      union { v8f v; float4 f[2]; } u; u.v = acc0;
      *(float4*)(dst)     = u.f[0];
      *(float4*)(dst + 4) = u.f[1];
    }
    __syncthreads();
    if (tid < 16) {
      float s = 0.f;
#pragma unroll
      for (int w = 0; w < NWAVE; ++w) s += redLds[w * 16 + tid];
      s += bias[t * 16 + tid];
      if (doTanh) s = tanhf(s);
      out[t * 16 + tid] = s;
    }
    __syncthreads();
  }
}

__device__ __forceinline__ void load_vec(bf16_t* vecLds, const float* __restrict__ src, int K) {
  __syncthreads();
  for (int i = threadIdx.x; i < K; i += BLOCK) vecLds[i] = (bf16_t)src[i];
  __syncthreads();
}

__device__ __forceinline__ void build_stage_vec(bf16_t* vecLds, const float* __restrict__ y,
                                                const float* __restrict__ kbuf, int s, float dt) {
  __syncthreads();
  for (int i = threadIdx.x; i < DIM; i += BLOCK) {
    float v = y[i];
#pragma unroll
    for (int j = 0; j < 5; ++j)
      if (j < s) v += dt * c_A[s][j] * kbuf[j * DIM + i];
    vecLds[i] = (bf16_t)v;
  }
  __syncthreads();
}

struct OdeParams {
  const bf16_t* W1; const bf16_t* W2; const bf16_t* W3;
  const float*  b1; const float*  b2; const float*  b3;
  const float*  z0;
  float* y; float* h1; float* h2; float* kbuf;
  unsigned* barCnt; unsigned* barGen;
  float* out;
  const int* t0; const int* t1; const int* ns;
};

__global__ __launch_bounds__(BLOCK) void ode_kernel(OdeParams p) {
  __shared__ __align__(16) bf16_t vecLds[HID];        // 8 KB
  __shared__ __align__(16) float  redLds[NWAVE * 16]; // 512 B

  const int   nsteps = *p.ns;
  const float dt = (float)(*p.t1 - *p.t0) / (float)nsteps;
  const int   gid = blockIdx.x * BLOCK + threadIdx.x;

  // traj row 0 = z0, y = z0
  if (gid < DIM) {
    float v = p.z0[gid];
    p.y[gid] = v;
    p.out[gid] = v;
  }
  grid_barrier(p.barCnt, p.barGen);

  for (int step = 0; step < nsteps - 1; ++step) {
    for (int s = 0; s < NSTAGE; ++s) {
      build_stage_vec(vecLds, p.y, p.kbuf, s, dt);
      matvec_layer(p.W1, p.b1, vecLds, redLds, p.h1, HID, DIM, true);
      grid_barrier(p.barCnt, p.barGen);

      load_vec(vecLds, p.h1, HID);
      matvec_layer(p.W2, p.b2, vecLds, redLds, p.h2, HID, HID, true);
      grid_barrier(p.barCnt, p.barGen);

      load_vec(vecLds, p.h2, HID);
      matvec_layer(p.W3, p.b3, vecLds, redLds, p.kbuf + s * DIM, DIM, HID, false);
      grid_barrier(p.barCnt, p.barGen);
    }
    if (gid < DIM) {
      const float* k = p.kbuf;
      float v = p.y[gid] + dt * (TB1 * k[gid] + TB2 * k[DIM + gid] + TB3 * k[2 * DIM + gid] +
                                 TB4 * k[3 * DIM + gid] + TB5 * k[4 * DIM + gid] +
                                 TB6 * k[5 * DIM + gid]);
      p.y[gid] = v;
      p.out[(size_t)(step + 1) * DIM + gid] = v;
    }
    grid_barrier(p.barCnt, p.barGen);
  }
}

extern "C" void kernel_launch(void* const* d_in, const int* in_sizes, int n_in,
                              void* d_out, int out_size, void* d_ws, size_t ws_size,
                              hipStream_t stream) {
  (void)in_sizes; (void)n_in; (void)out_size; (void)ws_size;
  const float* z0 = (const float*)d_in[0];
  const float* W1 = (const float*)d_in[1];
  const float* b1 = (const float*)d_in[2];
  const float* W2 = (const float*)d_in[3];
  const float* b2 = (const float*)d_in[4];
  const float* W3 = (const float*)d_in[5];
  const float* b3 = (const float*)d_in[6];
  const int*   t0 = (const int*)d_in[7];
  const int*   t1 = (const int*)d_in[8];
  const int*   ns = (const int*)d_in[9];

  char* ws = (char*)d_ws;
  bf16_t* W1p = (bf16_t*)(ws + OFF_W1);
  bf16_t* W2p = (bf16_t*)(ws + OFF_W2);
  bf16_t* W3p = (bf16_t*)(ws + OFF_W3);

  hipMemsetAsync(ws + OFF_BAR, 0, 256, stream);
  packW<<<(HID * DIM + 255) / 256, 256, 0, stream>>>(W1, W1p, HID, DIM);
  packW<<<(HID * HID + 255) / 256, 256, 0, stream>>>(W2, W2p, HID, HID);
  packW<<<(DIM * HID + 255) / 256, 256, 0, stream>>>(W3, W3p, DIM, HID);

  OdeParams p;
  p.W1 = W1p; p.W2 = W2p; p.W3 = W3p;
  p.b1 = b1;  p.b2 = b2;  p.b3 = b3;
  p.z0 = z0;
  p.y    = (float*)(ws + OFF_Y);
  p.h1   = (float*)(ws + OFF_H1);
  p.h2   = (float*)(ws + OFF_H2);
  p.kbuf = (float*)(ws + OFF_K);
  p.barCnt = (unsigned*)(ws + OFF_BAR);
  p.barGen = (unsigned*)(ws + OFF_BAR + 64);
  p.out = (float*)d_out;
  p.t0 = t0; p.t1 = t1; p.ns = ns;

  ode_kernel<<<NWG, BLOCK, 0, stream>>>(p);
}
